// Event_Memory_Unit_2551210574597
// MI455X (gfx1250) — matrix operand order repaired and checked by reading, not compiled, not run on hardware
//
#include <hip/hip_runtime.h>
#include <hip/hip_bf16.h>
#include <math.h>

// ---------------- types ----------------
typedef __attribute__((ext_vector_type(16))) __bf16 v16bf;
typedef __attribute__((ext_vector_type(8)))  __bf16 v8bf;
typedef __attribute__((ext_vector_type(8)))  float  v8f;
typedef __attribute__((ext_vector_type(4)))  unsigned int v4u;
typedef __attribute__((ext_vector_type(8)))  int v8i;
typedef __attribute__((ext_vector_type(4)))  int v4i;

union AFrag { v16bf v; v8bf h[2]; };

__device__ __forceinline__ __bf16 f2bf(float f) {
  union { float f; unsigned u; } x; x.f = f;
  unsigned r = (x.u + 0x7FFFu + ((x.u >> 16) & 1u)) >> 16;
  union { unsigned short s; __bf16 b; } y; y.s = (unsigned short)r;
  return y.b;
}
__device__ __forceinline__ float bf2f(__bf16 b) {
  union { unsigned short s; __bf16 b; } x; x.b = b;
  union { unsigned u; float f; } y; y.u = ((unsigned)x.s) << 16;
  return y.f;
}

#ifndef __has_builtin
#define __has_builtin(x) 0
#endif
#if __has_builtin(__builtin_amdgcn_tensor_load_to_lds) && __has_builtin(__builtin_amdgcn_s_wait_tensorcnt)
#define USE_TDM 1
#else
#define USE_TDM 0
#endif

#define TILE_M 64
#define TILE_N 64
#define KSTEP  32

// ------------- bf16 WMMA GEMM: C = A(bf16,MxK) * B'(bf16,[N][K]) -----------
// B' is pre-transposed so rows are K-contiguous (WMMA B-fragment layout).
// B panel (64 x KD) for this block is staged into LDS by one TDM DMA.
// epilogue: 0 = store f32, 1 = sigmoid(acc*scale)->bf16, 2 = +bias[col]->f32,
//           3 = store bf16
__global__ __launch_bounds__(256)
void wmma_gemm_kernel(const __bf16* __restrict__ A, int lda,
                      const __bf16* __restrict__ B, int ldb,
                      void* __restrict__ Cout, int ldc,
                      int KD, int Ntot, int epilogue,
                      const float* __restrict__ bias, float scale)
{
  extern __shared__ __align__(16) __bf16 Bs[];   // [64][KD], K-contiguous rows

  const int tid  = threadIdx.x;
  const int lane = tid & 31;
  const int wave = tid >> 5;
  const int rt   = wave & 3;   // row tile 0..3
  const int cp   = wave >> 2;  // col half 0..1 (two 16-col tiles each)
  const int m0   = blockIdx.x * TILE_M;
  const int n0   = blockIdx.y * TILE_N;

  // ---- stage this block's B panel: B'[n0..n0+63][0..KD) -> LDS ----
#if USE_TDM
  if (wave == 0) {
    const unsigned long long ga = (unsigned long long)(size_t)(B + (size_t)n0 * ldb);
    const unsigned lds_base = (unsigned)(size_t)&Bs[0];
    v4u g0;
    g0[0] = 1u;                                            // count=1
    g0[1] = lds_base;                                      // lds_addr
    g0[2] = (unsigned)(ga & 0xFFFFFFFFu);                  // global_addr lo
    g0[3] = (unsigned)(((ga >> 32) & 0x01FFFFFFu) | (2u << 30)); // addr hi | type=2
    v8i g1;
    g1[0] = (1 << 16);                                     // data_size=1 (2B), no mask
    g1[1] = (int)((KD & 0xFFFF) << 16);                    // tensor_dim0 lo16
    g1[2] = (int)(((KD >> 16) & 0xFFFF) | ((Ntot & 0xFFFF) << 16)); // dim0 hi | dim1 lo
    g1[3] = (int)(((Ntot >> 16) & 0xFFFF) | ((KD & 0xFFFF) << 16)); // dim1 hi | tile_dim0=KD
    g1[4] = 64;                                            // tile_dim1=64, tile_dim2=0
    g1[5] = ldb;                                           // tensor_dim0_stride lo32
    g1[6] = 0;
    g1[7] = 0;
    v4i gz  = {0, 0, 0, 0};
    v8i gz8 = {0, 0, 0, 0, 0, 0, 0, 0};
    __builtin_amdgcn_tensor_load_to_lds(g0, g1, gz, gz, gz8, 0);
    __builtin_amdgcn_s_wait_tensorcnt(0);
  }
#else
  for (int idx = tid; idx < 64 * KD; idx += 256)
    Bs[idx] = B[(size_t)(n0 + idx / KD) * ldb + (idx % KD)];
#endif
  __syncthreads();

  // fragment coordinates (ISA 7.12.2 layouts)
  const int mrow = m0 + rt * 16 + (lane & 15);
  const int kbA  = (lane >> 4) * 8;    // A: lanes 0-15 K=0..7/16..23; 16-31 K=8..15/24..31
  const int nloc = lane & 15;
  const int kbB  = (lane >> 4) * 8;    // B: lanes 0-15 K=0..7/16..23; 16-31 K=8..15/24..31
  const __bf16* Arow = A + (size_t)mrow * lda;
  const __bf16* Brow0 = &Bs[(size_t)(cp * 32 + nloc) * KD];
  const __bf16* Brow1 = &Bs[(size_t)(cp * 32 + 16 + nloc) * KD];

  v8f acc0 = {}; v8f acc1 = {};

  for (int k0 = 0; k0 < KD; k0 += 2 * KSTEP) {
    if (k0 + 2 * KSTEP < KD)
      __builtin_prefetch(Arow + k0 + 2 * KSTEP, 0, 0);   // global_prefetch_b8
    #pragma unroll
    for (int s = 0; s < 2; ++s) {
      const int kk = k0 + s * KSTEP;
      AFrag a;
      a.h[0] = *(const v8bf*)(Arow + kk + kbA);
      a.h[1] = *(const v8bf*)(Arow + kk + kbA + 16);
      AFrag b0, b1;
      b0.h[0] = *(const v8bf*)(Brow0 + kk + kbB);
      b0.h[1] = *(const v8bf*)(Brow0 + kk + kbB + 16);
      b1.h[0] = *(const v8bf*)(Brow1 + kk + kbB);
      b1.h[1] = *(const v8bf*)(Brow1 + kk + kbB + 16);
      acc0 = __builtin_amdgcn_wmma_f32_16x16x32_bf16(false, a.v, false, b0.v,
                                                     (short)0, acc0, false, false);
      acc1 = __builtin_amdgcn_wmma_f32_16x16x32_bf16(false, a.v, false, b1.v,
                                                     (short)0, acc1, false, false);
    }
  }

  // store (16x16 f32 C/D layout)
  const int mbase = rt * 16 + (lane >> 4) * 8;
  const int c0 = n0 + cp * 32 + (lane & 15);
  #pragma unroll
  for (int r = 0; r < 8; ++r) {
    const int grow = m0 + mbase + r;
    float v0 = acc0[r], v1 = acc1[r];
    const size_t i0 = (size_t)grow * ldc + c0;
    const size_t i1 = i0 + 16;
    if (epilogue == 1) {
      v0 = 1.0f / (1.0f + expf(-v0 * scale));
      v1 = 1.0f / (1.0f + expf(-v1 * scale));
      ((__bf16*)Cout)[i0] = f2bf(v0);
      ((__bf16*)Cout)[i1] = f2bf(v1);
    } else if (epilogue == 3) {
      ((__bf16*)Cout)[i0] = f2bf(v0);
      ((__bf16*)Cout)[i1] = f2bf(v1);
    } else if (epilogue == 2) {
      ((float*)Cout)[i0] = v0 + bias[c0];
      ((float*)Cout)[i1] = v1 + bias[c0 + 16];
    } else {
      ((float*)Cout)[i0] = v0;
      ((float*)Cout)[i1] = v1;
    }
  }
}

// ---------------- f32 -> bf16 convert / transpose ---------------------------
__global__ void f32_to_bf16_kernel(const float* __restrict__ src,
                                   __bf16* __restrict__ dst, size_t n)
{
  const size_t i = (size_t)blockIdx.x * blockDim.x + threadIdx.x;
  if (i < n) dst[i] = f2bf(src[i]);
}

// dst[c][r] = src[r][c] :  (R x C) f32 -> (C x R) bf16
__global__ void transpose_to_bf16_kernel(const float* __restrict__ src,
                                         __bf16* __restrict__ dst, int R, int C)
{
  const size_t i = (size_t)blockIdx.x * blockDim.x + threadIdx.x;
  if (i >= (size_t)R * C) return;
  const int r = (int)(i / C), c = (int)(i % C);
  dst[(size_t)c * R + r] = f2bf(src[i]);
}

// ---------------- t_att = mean(top5 of 64) per row --------------------------
__global__ void tatt_kernel(const __bf16* __restrict__ att, float* __restrict__ out_att)
{
  const int gid = blockIdx.x * blockDim.x + threadIdx.x;
  if (gid >= 2 * 65536) return;
  const int mem = gid >> 16;
  const int r   = gid & 65535;
  const __bf16* row = att + (size_t)gid * 64;
  float t[5] = {-1e30f, -1e30f, -1e30f, -1e30f, -1e30f};
  for (int i = 0; i < 64; ++i) {
    const float v = bf2f(row[i]);
    if (v > t[4]) {
      t[4] = v;
      #pragma unroll
      for (int j = 4; j > 0; --j)
        if (t[j] > t[j - 1]) { const float tmp = t[j]; t[j] = t[j - 1]; t[j - 1] = tmp; }
    }
  }
  const float s = (t[0] + t[1] + t[2] + t[3] + t[4]) * 0.2f;
  const int half = r >> 15, rr = r & 32767;
  const int slot = (mem == 0) ? (half ? 0 : 2) : (half ? 3 : 1); // A_att,N_att,A_Natt,N_Aatt
  out_att[(size_t)slot * 32768 + rr] = s;
}

// ---------------- fuse: reparam + F_M assembly ------------------------------
__global__ void fuse_kernel(const float* __restrict__ x, const float* __restrict__ eps,
                            const float* __restrict__ Y0, const float* __restrict__ Y1,
                            const float* __restrict__ Y2, const float* __restrict__ Y3,
                            const float* __restrict__ Y4,
                            float* __restrict__ Nnew, float* __restrict__ FM)
{
  const size_t i = (size_t)blockIdx.x * blockDim.x + threadIdx.x;
  if (i >= (size_t)16777216) return;
  const int d = (int)(i & 511);
  const size_t r = i >> 9;
  const float nv = Y0[i] + eps[i] * sqrtf(expf(Y1[i]));
  Nnew[i] = nv;
  FM[r * 1024 + d]        = x[i];
  FM[r * 1024 + 512 + d]  = nv + Y3[i];
  const size_t rH = r + (size_t)64 * 512;
  FM[rH * 1024 + d]       = x[i + (size_t)64 * 512 * 512];
  FM[rH * 1024 + 512 + d] = Y2[i] + Y4[i];
}

// ---------------- kl reduction ---------------------------------------------
__global__ void init_scalars_kernel(float* s) { s[0] = 0.f; s[1] = 0.f; s[2] = 0.f; }

__global__ void kl_kernel(const float* __restrict__ mu, const float* __restrict__ var,
                          float* __restrict__ out_kl)
{
  __shared__ float red[256];
  float s = 0.f;
  for (size_t i = (size_t)blockIdx.x * 256 + threadIdx.x; i < (size_t)16777216;
       i += (size_t)gridDim.x * 256) {
    const float m = mu[i], v = var[i];
    s += 1.0f + v - m * m - expf(v);
  }
  red[threadIdx.x] = s;
  __syncthreads();
  for (int o = 128; o > 0; o >>= 1) {
    if (threadIdx.x < o) red[threadIdx.x] += red[threadIdx.x + o];
    __syncthreads();
  }
  if (threadIdx.x == 0) atomicAdd(out_kl, red[0] * (-0.5f / 32768.0f));
}

// ---------------- top-33 select + gather means ------------------------------
__global__ void topk_gather_kernel(const float* __restrict__ attBase,
                                   const float* __restrict__ x,
                                   const float* __restrict__ Y2,
                                   const float* __restrict__ Nnew,
                                   float* __restrict__ vecs)
{
  __shared__ float sv[512];
  __shared__ int   sidx[33];
  const int b = blockIdx.x, task = blockIdx.y;
  const float* A_x = x + (size_t)64 * 512 * 512;
  const float* N_x = x;

  const float* att; const float* f1; const float* f2; float* d1; float* d2;
  if (task == 0)      { att = attBase;             f1 = A_x; f2 = Y2;      d1 = vecs + 0 * 32768; d2 = vecs + 4 * 32768; }
  else if (task == 1) { att = attBase + 1 * 32768; f1 = N_x; f2 = Nnew;    d1 = vecs + 1 * 32768; d2 = vecs + 3 * 32768; }
  else                { att = attBase + 3 * 32768; f1 = A_x; f2 = nullptr; d1 = vecs + 2 * 32768; d2 = nullptr; }

  for (int d = threadIdx.x; d < 512; d += blockDim.x) sv[d] = att[(size_t)b * 512 + d];
  __syncthreads();
  if (threadIdx.x == 0) {
    for (int j = 0; j < 33; ++j) {
      int bi = 0; float bv = -1e30f;
      for (int i = 0; i < 512; ++i) if (sv[i] > bv) { bv = sv[i]; bi = i; }
      sidx[j] = bi; sv[bi] = -1e30f;
    }
  }
  __syncthreads();
  for (int d = threadIdx.x; d < 512; d += blockDim.x) {
    float s1 = 0.f, s2 = 0.f;
    for (int j = 0; j < 33; ++j) {
      const size_t ro = ((size_t)b * 512 + sidx[j]) * 512 + d;
      s1 += f1[ro];
      if (f2) s2 += f2[ro];
    }
    d1[(size_t)b * 512 + d] = s1 * (1.0f / 33.0f);
    if (d2) d2[(size_t)b * 512 + d] = s2 * (1.0f / 33.0f);
  }
}

// ---------------- triplet + distance scalars --------------------------------
__global__ void loss_kernel(const float* __restrict__ vecs, float* __restrict__ out_sc)
{
  __shared__ float st[64], sd[64];
  const int b = threadIdx.x;  // 64 threads
  const float* n  = vecs + 0 * 32768 + (size_t)b * 512;
  const float* a  = vecs + 1 * 32768 + (size_t)b * 512;
  const float* p  = vecs + 2 * 32768 + (size_t)b * 512;
  const float* an = vecs + 3 * 32768 + (size_t)b * 512;
  const float* ng = vecs + 4 * 32768 + (size_t)b * 512;

  float na = 0.f, npp = 0.f, nn = 0.f, nan_ = 0.f, nng = 0.f;
  for (int d = 0; d < 512; ++d) {
    na  += a[d] * a[d];  npp += p[d] * p[d];  nn  += n[d] * n[d];
    nan_ += an[d] * an[d]; nng += ng[d] * ng[d];
  }
  na = sqrtf(na); npp = sqrtf(npp); nn = sqrtf(nn);
  float dap = 0.f, dan = 0.f;
  for (int d = 0; d < 512; ++d) {
    const float ea = a[d] / na;
    const float dp = ea - p[d] / npp + 1e-6f;
    const float dn = ea - n[d] / nn  + 1e-6f;
    dap += dp * dp; dan += dn * dn;
  }
  st[b] = fmaxf(sqrtf(dap) - sqrtf(dan) + 1.0f, 0.0f);
  sd[b] = fmaxf(100.0f - sqrtf(nng) + sqrtf(nan_), 0.0f);
  __syncthreads();
  for (int o = 32; o > 0; o >>= 1) {
    if (b < o) { st[b] += st[b + o]; sd[b] += sd[b + o]; }
    __syncthreads();
  }
  if (b == 0) { out_sc[0] = st[0] / 64.0f; out_sc[2] = sd[0] / 64.0f; }
}

// ---------------- host ------------------------------------------------------
extern "C" void kernel_launch(void* const* d_in, const int* in_sizes, int n_in,
                              void* d_out, int out_size, void* d_ws, size_t ws_size,
                              hipStream_t stream)
{
  const float* x     = (const float*)d_in[0];
  const float* Amem  = (const float*)d_in[1];
  const float* Nmem  = (const float*)d_in[2];
  const float* W_mu  = (const float*)d_in[3];
  const float* b_mu  = (const float*)d_in[4];
  const float* W_var = (const float*)d_in[5];
  const float* b_var = (const float*)d_in[6];
  const float* eps   = (const float*)d_in[7];
  float* out = (float*)d_out;

  const size_t FM_ELEMS = (size_t)128 * 512 * 1024;
  float* out_scalars = out + FM_ELEMS;        // [triplet, kl, distance]
  float* out_att     = out + FM_ELEMS + 3;    // A_att, N_att, A_Natt, N_Aatt

  // ---- workspace carve-up (bytes) ----
  char* p = (char*)d_ws;
  const size_t M1 = 65536, M2 = 32768;
  __bf16* xbf   = (__bf16*)p; p += M1 * 512 * 2;            // 64 MB
  __bf16* mem0b = (__bf16*)p; p += 64 * 512 * 2;            // mem rows, [N=64][K=512]
  __bf16* mem1b = (__bf16*)p; p += 64 * 512 * 2;
  __bf16* memT0 = (__bf16*)p; p += 512 * 64 * 2;            // [N=512][K=64]
  __bf16* memT1 = (__bf16*)p; p += 512 * 64 * 2;
  __bf16* WmuT  = (__bf16*)p; p += 512 * 512 * 2;           // [N=512][K=512]
  __bf16* WvarT = (__bf16*)p; p += 512 * 512 * 2;
  __bf16* att0b = (__bf16*)p; p += M1 * 64 * 2;
  __bf16* att1b = (__bf16*)p; p += M1 * 64 * 2;
  __bf16* aug0b = (__bf16*)p; p += M1 * 512 * 2;
  __bf16* aug1b = (__bf16*)p; p += M1 * 512 * 2;
  float*  Y0 = (float*)p; p += M2 * 512 * 4;
  float*  Y1 = (float*)p; p += M2 * 512 * 4;
  float*  Y2 = (float*)p; p += M2 * 512 * 4;
  float*  Y3 = (float*)p; p += M2 * 512 * 4;
  float*  Y4 = (float*)p; p += M2 * 512 * 4;
  float*  Nnew = (float*)p; p += M2 * 512 * 4;
  float*  vecs = (float*)p;

  const dim3 blk(256);
  const float invs = 0.044194173824159216f; // 1/sqrt(512)

  // ---- one-time bf16 materialization (amortized across all GEMM reads) ----
  f32_to_bf16_kernel<<<(unsigned)((M1 * 512 + 255) / 256), blk, 0, stream>>>(x, xbf, M1 * 512);
  f32_to_bf16_kernel<<<128, blk, 0, stream>>>(Amem, mem0b, 64 * 512);
  f32_to_bf16_kernel<<<128, blk, 0, stream>>>(Nmem, mem1b, 64 * 512);
  transpose_to_bf16_kernel<<<128, blk, 0, stream>>>(Amem, memT0, 64, 512);
  transpose_to_bf16_kernel<<<128, blk, 0, stream>>>(Nmem, memT1, 64, 512);
  transpose_to_bf16_kernel<<<1024, blk, 0, stream>>>(W_mu,  WmuT,  512, 512);
  transpose_to_bf16_kernel<<<1024, blk, 0, stream>>>(W_var, WvarT, 512, 512);

  // ---- attention logits + sigmoid -> bf16 att ----
  wmma_gemm_kernel<<<dim3(1024, 1), blk, 64 * 512 * 2, stream>>>(
      xbf, 512, mem0b, 512, att0b, 64, 512, 64, 1, nullptr, invs);
  wmma_gemm_kernel<<<dim3(1024, 1), blk, 64 * 512 * 2, stream>>>(
      xbf, 512, mem1b, 512, att1b, 64, 512, 64, 1, nullptr, invs);

  // t_att (top-5 mean) -> 4 att outputs (f32)
  tatt_kernel<<<512, blk, 0, stream>>>(att0b, out_att);

  // ---- aug = att @ mem -> bf16 ----
  wmma_gemm_kernel<<<dim3(1024, 8), blk, 64 * 64 * 2, stream>>>(
      att0b, 64, memT0, 64, aug0b, 512, 64, 512, 3, nullptr, 1.f);
  wmma_gemm_kernel<<<dim3(1024, 8), blk, 64 * 64 * 2, stream>>>(
      att1b, 64, memT1, 64, aug1b, 512, 64, 512, 3, nullptr, 1.f);

  // ---- projections (M=32768, bias epilogue, f32 out) ----
  wmma_gemm_kernel<<<dim3(512, 8), blk, 64 * 512 * 2, stream>>>(
      aug1b,            512, WmuT,  512, Y0, 512, 512, 512, 2, b_mu,  1.f); // N_aug_mu
  wmma_gemm_kernel<<<dim3(512, 8), blk, 64 * 512 * 2, stream>>>(
      aug1b,            512, WvarT, 512, Y1, 512, 512, 512, 2, b_var, 1.f); // N_aug_var
  wmma_gemm_kernel<<<dim3(512, 8), blk, 64 * 512 * 2, stream>>>(
      aug0b + M2 * 512, 512, WmuT,  512, Y2, 512, 512, 512, 2, b_mu,  1.f); // A_aug_new
  wmma_gemm_kernel<<<dim3(512, 8), blk, 64 * 512 * 2, stream>>>(
      aug0b,            512, WmuT,  512, Y3, 512, 512, 512, 2, b_mu,  1.f); // A_Naug_mu
  wmma_gemm_kernel<<<dim3(512, 8), blk, 64 * 512 * 2, stream>>>(
      aug1b + M2 * 512, 512, WmuT,  512, Y4, 512, 512, 512, 2, b_mu,  1.f); // N_Aaug_mu

  // ---- reparam + F_M assembly ----
  fuse_kernel<<<65536, blk, 0, stream>>>(x, eps, Y0, Y1, Y2, Y3, Y4, Nnew, out);

  // ---- scalars ----
  init_scalars_kernel<<<1, 1, 0, stream>>>(out_scalars);
  kl_kernel<<<2048, blk, 0, stream>>>(Y0, Y1, out_scalars + 1);

  // ---- top-33 + gather means, then triplet/distance ----
  topk_gather_kernel<<<dim3(64, 3), blk, 0, stream>>>(out_att, x, Y2, Nnew, vecs);
  loss_kernel<<<1, 64, 0, stream>>>(vecs, out_scalars);

  (void)in_sizes; (void)n_in; (void)out_size; (void)ws_size;
}
